// MyRNN_56186762167074
// MI455X (gfx1250) — matrix operand order repaired and checked by reading, not compiled, not run on hardware
//
#include <hip/hip_runtime.h>
#include <hip/hip_bf16.h>
#include <math.h>

// ---------------------------------------------------------------------------
// 2-layer GRU (Keras reset_after=true) for MI455X / gfx1250, bf16 WMMA path.
//
// B=1024, T=80, U=512, D=100(embed, padded to 128), 3U=1536.
// Per kernel_launch:
//   1. transpose+convert 4 weight matrices to bf16 W^T  (parallel, tiny)
//   2. embedding gather -> bf16 x[t][b][128]            (parallel, 21 MB)
//   3. zero-init h ping-pong buffers (f32 master + bf16 GEMM copy)
//   4. 80 x { gru_step(layer1,t); gru_step(layer2,t) }  (sequential chain)
//   5. fc head: sigmoid(h_last @ Wfc + bfc), wave-per-row reduction
// gru_step fuses BOTH GEMMs (x-path + h-path, 3 gates each) and the full
// GRU gating using v_wmma_f32_16x16x32_bf16. Each wave owns a 32x16 tile
// (two M sub-tiles share the three B gate fragments -> fewer loads/WMMA).
// ---------------------------------------------------------------------------

#define GRU_B 1024
#define GRU_T 80
#define GRU_U 512
#define GRU_KX1 128   // embed dim 100 padded to 128 (zeros)
#define GRU_3U 1536

typedef __attribute__((ext_vector_type(16))) __bf16 bf16x16;
typedef __attribute__((ext_vector_type(8)))  __bf16 bf16x8;
typedef __attribute__((ext_vector_type(8)))  float  f32x8;

// ---- WMMA fragment loaders (contiguous k = 16h + s, same order in A and B) --------

// A (16x32 bf16, row-major activations [.][lda]):
// lanes 0-15 : elems 0-15 -> K0..15   (row M = lane)
// lanes 16-31: elems 0-15 -> K16..31  (row M = lane-16)
__device__ __forceinline__ bf16x16
load_a_frag(const __bf16* __restrict__ base, int lda, int mrow, int k0, int lane) {
    int off = (lane & 16) ? 16 : 0;
    const __bf16* p = base + (size_t)mrow * lda + k0;
    bf16x8 lo = *(const bf16x8*)(p + off);
    bf16x8 hi = *(const bf16x8*)(p + 8 + off);
    return __builtin_shufflevector(lo, hi, 0,1,2,3,4,5,6,7,8,9,10,11,12,13,14,15);
}

// B (32x16 bf16) from pre-transposed weights WT[N][ldk] (contiguous in K):
// lanes 0-15 : elems 0-15 -> K0..15 of column N=lane
// lanes 16-31: elems 0-15 -> K16..31 of column N=lane-16
__device__ __forceinline__ bf16x16
load_b_frag(const __bf16* __restrict__ wt, int ldk, int nrow, int k0, int lane) {
    int off = (lane & 16) ? 16 : 0;
    return *(const bf16x16*)(wt + (size_t)nrow * ldk + k0 + off);
}

__device__ __forceinline__ f32x8 wmma_bf16(bf16x16 a, bf16x16 b, f32x8 c) {
    return __builtin_amdgcn_wmma_f32_16x16x32_bf16(
        /*neg_a=*/false, a, /*neg_b=*/false, b,
        /*c_mod=*/(short)0, c, /*reuse_a=*/false, /*reuse_b=*/false);
}

// ---- Fused GRU step --------------------------------------------------------
// Grid: (8, 16) = (M tiles of 128 over B=1024, N tiles of 32 over U=512)
// Block: 256 threads = 8 waves; wave (wm,wn) owns a 32x16 tile (2 M subtiles).
__global__ __launch_bounds__(256) void gru_step_kernel(
    const __bf16* __restrict__ X,   int Kx,                 // [B][Kx] bf16
    const __bf16* __restrict__ WxT,                          // [3U][Kx] bf16
    const __bf16* __restrict__ Hbf,                          // [B][U] bf16 (h_prev)
    const __bf16* __restrict__ WhT,                          // [3U][U] bf16
    const float*  __restrict__ bias,                         // [2][3U] f32 (bi, br)
    const float*  __restrict__ Hprev,                        // [B][U] f32
    float*        __restrict__ Hnew,                         // [B][U] f32
    __bf16*       __restrict__ Hnewbf)                       // [B][U] bf16
{
    const int U = GRU_U;
    const int lane = threadIdx.x & 31;
    const int w    = threadIdx.x >> 5;
    const int wm   = w & 3;          // 0..3 -> 32-row group
    const int wn   = w >> 2;         // 0..1 -> 16-col group
    const int m0   = blockIdx.x * 128 + wm * 32;
    const int n0   = blockIdx.y * 32 + wn * 16;
    const int mrow = m0 + (lane & 15);       // sub-tile 0 row; sub-tile 1 = +16
    const int ncol = n0 + (lane & 15);

    f32x8 ax[3][2];   // [gate z/r/n][m-subtile]  x-path
    f32x8 ah[3][2];   //                          h-path
#pragma unroll
    for (int g = 0; g < 3; ++g)
#pragma unroll
        for (int s = 0; s < 2; ++s) { ax[g][s] = (f32x8){}; ah[g][s] = (f32x8){}; }

    // gh = h_prev @ Wh  (K = 512): 3 B frags shared by 2 A frags -> 6 WMMA / 10 loads
    for (int k0 = 0; k0 < U; k0 += 32) {
        bf16x16 a0 = load_a_frag(Hbf, U, mrow,      k0, lane);
        bf16x16 a1 = load_a_frag(Hbf, U, mrow + 16, k0, lane);
#pragma unroll
        for (int g = 0; g < 3; ++g) {
            bf16x16 bg = load_b_frag(WhT, U, ncol + 512 * g, k0, lane);
            ah[g][0] = wmma_bf16(a0, bg, ah[g][0]);
            ah[g][1] = wmma_bf16(a1, bg, ah[g][1]);
        }
    }
    // gx = x @ Wx  (K = 128 for layer1, 512 for layer2)
    for (int k0 = 0; k0 < Kx; k0 += 32) {
        bf16x16 a0 = load_a_frag(X, Kx, mrow,      k0, lane);
        bf16x16 a1 = load_a_frag(X, Kx, mrow + 16, k0, lane);
#pragma unroll
        for (int g = 0; g < 3; ++g) {
            bf16x16 bg = load_b_frag(WxT, Kx, ncol + 512 * g, k0, lane);
            ax[g][0] = wmma_bf16(a0, bg, ax[g][0]);
            ax[g][1] = wmma_bf16(a1, bg, ax[g][1]);
        }
    }

    // GRU gating. C/D layout: VGPR i holds row M = base + i + 8*(lane>=16), col N = ncol.
    const float biz = bias[ncol],        bir = bias[512 + ncol],  bih = bias[1024 + ncol];
    const float brz = bias[1536 + ncol], brr = bias[2048 + ncol], brn = bias[2560 + ncol];
#pragma unroll
    for (int s = 0; s < 2; ++s) {
        const int mbase = m0 + s * 16 + ((lane & 16) ? 8 : 0);
#pragma unroll
        for (int i = 0; i < 8; ++i) {
            const int m = mbase + i;
            const float xz = ax[0][s][i] + biz, xr = ax[1][s][i] + bir, xh = ax[2][s][i] + bih;
            const float hz = ah[0][s][i] + brz, hr = ah[1][s][i] + brr, hn = ah[2][s][i] + brn;
            const float z  = 1.0f / (1.0f + __expf(-(xz + hz)));
            const float r  = 1.0f / (1.0f + __expf(-(xr + hr)));
            const float hh = tanhf(xh + r * hn);
            const float hp = Hprev[(size_t)m * U + ncol];
            const float hv = z * hp + (1.0f - z) * hh;
            Hnew[(size_t)m * U + ncol]   = hv;
            Hnewbf[(size_t)m * U + ncol] = (__bf16)hv;
        }
    }
}

// ---- Aux kernels -----------------------------------------------------------

// dst[n][k] = (k < Ksrc) ? src[k][n] : 0   (f32 [Ksrc][1536] -> bf16 [1536][Kpad])
__global__ void transpose_convert_kernel(const float* __restrict__ src,
                                         __bf16* __restrict__ dst,
                                         int Ksrc, int Kpad) {
    int idx = blockIdx.x * blockDim.x + threadIdx.x;
    int total = GRU_3U * Kpad;
    if (idx >= total) return;
    int k = idx % Kpad;
    int n = idx / Kpad;
    float v = (k < Ksrc) ? src[(size_t)k * GRU_3U + n] : 0.0f;
    dst[idx] = (__bf16)v;
}

// xemb[t][b][k] = (k < 100) ? emb[tokens[b][t]][k] : 0   (bf16, Kpad=128)
__global__ void embed_gather_kernel(const int* __restrict__ tokens,
                                    const float* __restrict__ emb,
                                    __bf16* __restrict__ xemb) {
    int idx = blockIdx.x * blockDim.x + threadIdx.x;
    if (idx >= GRU_T * GRU_B * GRU_KX1) return;
    int k  = idx & (GRU_KX1 - 1);
    int rb = idx >> 7;               // t*B + b
    int b  = rb & (GRU_B - 1);
    int t  = rb >> 10;
    float v = 0.0f;
    if (k < 100) {
        int tok = tokens[(size_t)b * GRU_T + t];
        v = emb[(size_t)tok * 100 + k];
    }
    xemb[idx] = (__bf16)v;
}

__global__ void zero_f32_kernel(float* __restrict__ p, int n) {
    int i = blockIdx.x * blockDim.x + threadIdx.x;
    if (i < n) p[i] = 0.0f;
}
__global__ void zero_bf16_kernel(__bf16* __restrict__ p, int n) {
    int i = blockIdx.x * blockDim.x + threadIdx.x;
    if (i < n) p[i] = (__bf16)0.0f;
}

// out[b] = sigmoid(h[b][:] . Wfc + bfc). One wave32 per batch row.
__global__ __launch_bounds__(256) void final_fc_kernel(
    const float* __restrict__ h, const float* __restrict__ Wfc,
    const float* __restrict__ bfc, float* __restrict__ out) {
    const int lane = threadIdx.x & 31;
    const int row  = blockIdx.x * 8 + (threadIdx.x >> 5);
    if (row >= GRU_B) return;
    float acc = 0.0f;
    for (int k = lane; k < GRU_U; k += 32)
        acc += h[(size_t)row * GRU_U + k] * Wfc[k];
#pragma unroll
    for (int m = 16; m >= 1; m >>= 1)
        acc += __shfl_xor(acc, m, 32);
    if (lane == 0)
        out[row] = 1.0f / (1.0f + __expf(-(acc + bfc[0])));
}

// ---- Host orchestration ----------------------------------------------------

extern "C" void kernel_launch(void* const* d_in, const int* in_sizes, int n_in,
                              void* d_out, int out_size, void* d_ws, size_t ws_size,
                              hipStream_t stream) {
    const int*   tokens = (const int*)  d_in[0];
    const float* emb    = (const float*)d_in[1];
    const float* Wx1    = (const float*)d_in[2];
    const float* Wh1    = (const float*)d_in[3];
    const float* b1     = (const float*)d_in[4];
    const float* Wx2    = (const float*)d_in[5];
    const float* Wh2    = (const float*)d_in[6];
    const float* b2     = (const float*)d_in[7];
    const float* Wfc    = (const float*)d_in[8];
    const float* bfc    = (const float*)d_in[9];
    float* out          = (float*)d_out;

    // Workspace carve-up (~39 MB total), 256B-aligned slices.
    char* ws = (char*)d_ws;
    size_t off = 0;
    auto carve = [&](size_t bytes) -> char* {
        char* p = ws + off;
        off = (off + bytes + 255) & ~(size_t)255;
        return p;
    };
    __bf16* WxT1 = (__bf16*)carve((size_t)GRU_3U * GRU_KX1 * 2);
    __bf16* WhT1 = (__bf16*)carve((size_t)GRU_3U * GRU_U   * 2);
    __bf16* WxT2 = (__bf16*)carve((size_t)GRU_3U * GRU_U   * 2);
    __bf16* WhT2 = (__bf16*)carve((size_t)GRU_3U * GRU_U   * 2);
    __bf16* xemb = (__bf16*)carve((size_t)GRU_T * GRU_B * GRU_KX1 * 2);
    const size_t HN = (size_t)GRU_B * GRU_U;
    float*  h1f[2]; __bf16* h1b[2]; float* h2f[2]; __bf16* h2b[2];
    for (int i = 0; i < 2; ++i) h1f[i] = (float*) carve(HN * 4);
    for (int i = 0; i < 2; ++i) h1b[i] = (__bf16*)carve(HN * 2);
    for (int i = 0; i < 2; ++i) h2f[i] = (float*) carve(HN * 4);
    for (int i = 0; i < 2; ++i) h2b[i] = (__bf16*)carve(HN * 2);

    const int TPB = 256;
    auto blocks = [&](size_t n) { return (unsigned)((n + TPB - 1) / TPB); };

    // 1. Weights -> bf16 W^T (padded K for layer-1 x weights)
    transpose_convert_kernel<<<blocks((size_t)GRU_3U * GRU_KX1), TPB, 0, stream>>>(Wx1, WxT1, 100, GRU_KX1);
    transpose_convert_kernel<<<blocks((size_t)GRU_3U * GRU_U),   TPB, 0, stream>>>(Wh1, WhT1, GRU_U, GRU_U);
    transpose_convert_kernel<<<blocks((size_t)GRU_3U * GRU_U),   TPB, 0, stream>>>(Wx2, WxT2, GRU_U, GRU_U);
    transpose_convert_kernel<<<blocks((size_t)GRU_3U * GRU_U),   TPB, 0, stream>>>(Wh2, WhT2, GRU_U, GRU_U);

    // 2. Embedding gather -> bf16 [T][B][128]
    embed_gather_kernel<<<blocks((size_t)GRU_T * GRU_B * GRU_KX1), TPB, 0, stream>>>(tokens, emb, xemb);

    // 3. h0 = 0 (only parity-0 buffers are read before being written)
    zero_f32_kernel <<<blocks(HN), TPB, 0, stream>>>(h1f[0], (int)HN);
    zero_f32_kernel <<<blocks(HN), TPB, 0, stream>>>(h2f[0], (int)HN);
    zero_bf16_kernel<<<blocks(HN), TPB, 0, stream>>>(h1b[0], (int)HN);
    zero_bf16_kernel<<<blocks(HN), TPB, 0, stream>>>(h2b[0], (int)HN);

    // 4. Recurrent chain: 80 x (layer1 step, layer2 step), ping-pong h buffers.
    dim3 grid(GRU_B / 128, GRU_U / 32);   // (8, 16)
    for (int t = 0; t < GRU_T; ++t) {
        int p = t & 1, q = p ^ 1;
        gru_step_kernel<<<grid, TPB, 0, stream>>>(
            xemb + (size_t)t * GRU_B * GRU_KX1, GRU_KX1, WxT1,
            h1b[p], WhT1, b1, h1f[p], h1f[q], h1b[q]);
        gru_step_kernel<<<grid, TPB, 0, stream>>>(
            h1b[q], GRU_U, WxT2,
            h2b[p], WhT2, b2, h2f[p], h2f[q], h2b[q]);
    }

    // 5. Head: after t=79 (odd), final h2 lives in parity-0 buffer.
    final_fc_kernel<<<(GRU_B + 7) / 8 / 1, TPB, 0, stream>>>(h2f[0], Wfc, bfc, out);
}
